// ProtoClassifier_74517682585681
// MI455X (gfx1250) — matrix operand order repaired and checked by reading, not compiled, not run on hardware
//
#include <hip/hip_runtime.h>
#include <hip/hip_bf16.h>
#include <math.h>

// Problem constants (from reference)
#define N_SAMP 32768
#define P_DIM  512
#define M_PROTO 2048
#define K_CLS  1000
#define K_PAD  1008          // K rounded up to multiple of 16 (padded classes masked)
#define ROWS_PER_BLK 64      // row stripe held in LDS as bf16
#define MSTRIDE 2056         // 2048 + 8 bf16 pad -> LDS bank-conflict-free row stride

// Stage-B W staging through LDS via the Tensor Data Mover
#define CHUNK_M 512                         // contraction chunk per TDM transfer
#define NCHUNK (M_PROTO / CHUNK_M)          // 4
#define NTILES (K_PAD / 16)                 // 63 class tiles
#define TOTAL_CHUNKS (NTILES * NCHUNK)      // 252
#define WROW_BYTES (CHUNK_M * 2 + 32)       // 1056: TDM pads +8 dwords per row
#define WBUF_BYTES (16 * WROW_BYTES)        // 16896 per buffer
#define DLDS_BYTES ((ROWS_PER_BLK) * (MSTRIDE) * 2)   // 263168
#define WOFF (DLDS_BYTES)                   // W buffers start here in LDS
#define COLMIN_OFF (WOFF + 2 * WBUF_BYTES)  // colmin after the two W buffers
#define SMEM_TOTAL (COLMIN_OFF + M_PROTO * 4)         // 305152 <= 320KB

typedef __attribute__((ext_vector_type(16))) __bf16 v16bf;
typedef __attribute__((ext_vector_type(8)))  __bf16 v8bf;
typedef __attribute__((ext_vector_type(8)))  float  v8f;
typedef __attribute__((ext_vector_type(4)))  unsigned int v4u;
typedef __attribute__((ext_vector_type(8)))  int v8i;
typedef __attribute__((ext_vector_type(4)))  int v4i;

// A-operand fragment per CDNA5 16-bit A layout: lanes 0-15 hold K={0..7,16..23},
// lanes 16-31 hold K={8..15,24..31} -> two contiguous 16B chunks per lane.
static __device__ __forceinline__ v16bf load_a_frag(const __bf16* base) {
  v8bf lo = *(const v8bf*)(base);
  v8bf hi = *(const v8bf*)(base + 16);
  v16bf r;
#pragma unroll
  for (int j = 0; j < 8; ++j) { r[j] = lo[j]; r[8 + j] = hi[j]; }
  return r;
}

static __device__ __forceinline__ v8f wmma_bf16(v16bf a, v16bf b, v8f c) {
  return __builtin_amdgcn_wmma_f32_16x16x32_bf16(false, a, false, b, (short)0, c,
                                                 false, false);
}

// Issue one TDM transfer: 16 rows x 512 bf16 tile of W (row stride M_PROTO) ->
// LDS at byte offset lds_off, with +8dword padding per row (bank spread +
// keeps 32B alignment for v16bf reads).  D# per CDNA5 ISA ch.8; 6-arg builtin
// form (amdgpu-toolchain clang-23): groups 2/3 unused for a 2-D tile.
static __device__ __forceinline__ void tdm_issue_w(unsigned lds_off,
                                                   const __bf16* gptr) {
  unsigned long long ga = (unsigned long long)(size_t)gptr;
  v4u g0;
  g0[0] = 1u;                                   // count=1, user descriptor
  g0[1] = lds_off;                              // LDS byte address
  g0[2] = (unsigned)(ga & 0xffffffffull);       // global_addr[31:0]
  g0[3] = (unsigned)((ga >> 32) & 0x01ffffffull) | (2u << 30);  // addr hi,type=2
  const unsigned td0 = M_PROTO;                 // tensor dim0 (elements)
  const unsigned td1 = K_PAD;                   // tensor dim1 (rows)
  v8i g1;
  // data_size=2B (bit16), pad_enable (bit20), pad_interval=7 (256 dwords,
  // bits 24:22), pad_amount=7 (8 dwords, bits 31:25)
  g1[0] = (int)((1u << 16) | (1u << 20) | (7u << 22) | (7u << 25));
  g1[1] = (int)((td0 & 0xffffu) << 16);                    // tensor_dim0 lo
  g1[2] = (int)((td0 >> 16) | ((td1 & 0xffffu) << 16));    // dim0 hi | dim1 lo
  g1[3] = (int)((td1 >> 16) | ((unsigned)CHUNK_M << 16));  // dim1 hi | tile_dim0
  g1[4] = (int)(16u);                                      // tile_dim1=16
  g1[5] = (int)(M_PROTO);                                  // tensor_dim0_stride
  g1[6] = 0;
  g1[7] = 0;
  v4i z4 = {0, 0, 0, 0};
  v8i z8 = {0, 0, 0, 0, 0, 0, 0, 0};
  __builtin_amdgcn_tensor_load_to_lds(g0, g1, z4, z4, z8, 0);
}

// ---------------- prep kernels ----------------

// one wave per row: squared norm + bf16 convert
__global__ __launch_bounds__(256) void k_row_prep(const float* __restrict__ in,
                                                  __bf16* __restrict__ obf,
                                                  float* __restrict__ sq,
                                                  int rows, int cols) {
  int wave = (int)((blockIdx.x * blockDim.x + threadIdx.x) >> 5);
  int lane = threadIdx.x & 31;
  if (wave >= rows) return;
  const float* r = in + (size_t)wave * cols;
  __bf16* o = obf + (size_t)wave * cols;
  float s = 0.f;
  for (int c = lane; c < cols; c += 32) {
    float v = r[c];
    s += v * v;
    o[c] = (__bf16)v;
  }
#pragma unroll
  for (int off = 16; off > 0; off >>= 1) s += __shfl_xor(s, off, 32);
  if (lane == 0) sq[wave] = s;
}

// W [K,M] f32 -> bf16, zero-padded rows K..K_PAD
__global__ __launch_bounds__(256) void k_w_prep(const float* __restrict__ W,
                                                __bf16* __restrict__ wbf) {
  int idx = blockIdx.x * blockDim.x + threadIdx.x;
  if (idx >= K_PAD * M_PROTO) return;
  int k = idx / M_PROTO;
  int m = idx - k * M_PROTO;
  wbf[idx] = (k < K_CLS) ? (__bf16)W[(size_t)k * M_PROTO + m] : (__bf16)0.0f;
}

__global__ __launch_bounds__(256) void k_init(float* __restrict__ r1min,
                                              float* __restrict__ accum) {
  int t = blockIdx.x * blockDim.x + threadIdx.x;
  for (int c = t; c < M_PROTO; c += gridDim.x * blockDim.x) r1min[c] = 3.0e38f;
  if (t < 2) accum[t] = 0.f;
}

// ---------------- fused main kernel ----------------
// grid: N/64 blocks, 128 threads (4 waves; wave w owns rows [blk*64+w*16, +16)).
// Stage A: D stripe (bf16) -> LDS via WMMA, track r1/r2 mins.
// Stage B: logits = D @ W^T by WMMA, W staged via TDM, streaming logsumexp.
__global__ __launch_bounds__(128) void k_main(
    const __bf16* __restrict__ xbf, const __bf16* __restrict__ pbf,
    const __bf16* __restrict__ wbf, const float* __restrict__ x2,
    const float* __restrict__ p2, const int* __restrict__ y,
    float* __restrict__ r1min, float* __restrict__ accum) {
  extern __shared__ char smem_raw[];
  __bf16* Dlds = (__bf16*)smem_raw;
  char* wstage = smem_raw + WOFF;
  float* colmin = (float*)(smem_raw + COLMIN_OFF);

  const int lane = threadIdx.x & 31;
  const int w = threadIdx.x >> 5;     // wave id 0..3
  const int half = lane >> 4;         // lane half (C-tile row group / K chunk)
  const int l16 = lane & 15;
  const int rows0 = blockIdx.x * ROWS_PER_BLK;
  const int rbase = rows0 + w * 16;

  for (int c = threadIdx.x; c < M_PROTO; c += blockDim.x) colmin[c] = 3.0e38f;
  __syncthreads();

  // ---- Stage A: D = sqrt(relu(x2 + p2 - 2 x.p)) ----
  // Preload the wave's 16 x-rows for the whole P=512 contraction (16 A frags).
  v16bf afrag[16];
  {
    const __bf16* xrow = xbf + (size_t)(rbase + l16) * P_DIM;
#pragma unroll
    for (int kk = 0; kk < 16; ++kk)
      afrag[kk] = load_a_frag(xrow + kk * 32 + half * 8);
  }
  float x2v[8];
#pragma unroll
  for (int i = 0; i < 8; ++i) x2v[i] = x2[rbase + i + half * 8];

  float rmin[8];
#pragma unroll
  for (int i = 0; i < 8; ++i) rmin[i] = 3.0e38f;

  for (int mt = 0; mt < M_PROTO / 16; ++mt) {
    const int mcol = mt * 16 + l16;
    const __bf16* pcol = pbf + (size_t)mcol * P_DIM;  // column mcol of proto^T
    v8f acc = {0.f, 0.f, 0.f, 0.f, 0.f, 0.f, 0.f, 0.f};
#pragma unroll
    for (int kk = 0; kk < 16; ++kk) {
      v16bf b = load_a_frag(pcol + kk * 32 + half * 8);  // same K order as A
      acc = wmma_bf16(afrag[kk], b, acc);
    }
    const float p2c = p2[mcol];
    float cmin = 3.0e38f;
#pragma unroll
    for (int i = 0; i < 8; ++i) {
      float d2 = x2v[i] + p2c - 2.0f * acc[i];
      float d = sqrtf(fmaxf(d2, 0.0f));
      rmin[i] = fminf(rmin[i], d);
      cmin = fminf(cmin, d);
      Dlds[(size_t)(w * 16 + i + half * 8) * MSTRIDE + mt * 16 + l16] = (__bf16)d;
    }
    // column (prototype) min: combine the two lane halves, then LDS atomic.
    cmin = fminf(cmin, __shfl_xor(cmin, 16, 32));
    atomicMin((unsigned int*)&colmin[mcol], __float_as_uint(cmin));  // d >= 0
  }
  __syncthreads();

  // r1: flush block-local column mins to global
  for (int c = threadIdx.x; c < M_PROTO; c += blockDim.x)
    atomicMin((unsigned int*)&r1min[c], __float_as_uint(colmin[c]));

  // r2: finish per-row min (reduce across the 16 column lanes) and accumulate
#pragma unroll
  for (int i = 0; i < 8; ++i) {
#pragma unroll
    for (int off = 1; off < 16; off <<= 1)
      rmin[i] = fminf(rmin[i], __shfl_xor(rmin[i], off, 16));
  }
  if (l16 == 0) {
    float s = 0.f;
#pragma unroll
    for (int i = 0; i < 8; ++i) s += rmin[i];
    atomicAdd(&accum[1], s);
  }

  // ---- Stage B: logits = D @ W^T with streaming logsumexp ----
  float mrun[8], srun[8], ly[8];
  int yv[8];
#pragma unroll
  for (int i = 0; i < 8; ++i) {
    mrun[i] = -3.0e38f;
    srun[i] = 0.f;
    ly[i] = 0.f;
    yv[i] = y[rbase + i + half * 8];
  }

  const __bf16* arow = Dlds + (size_t)(w * 16 + l16) * MSTRIDE;

  // Prefetch first W chunk via TDM (wave 0 drives the DMA for the block).
  if (w == 0) tdm_issue_w((unsigned)WOFF, wbf);

  int t = 0;  // linear chunk index: t = kt*NCHUNK + c
  for (int kt = 0; kt < NTILES; ++kt) {
    const int kcol = kt * 16 + l16;
    const bool valid = (kcol < K_CLS);
    v8f acc = {0.f, 0.f, 0.f, 0.f, 0.f, 0.f, 0.f, 0.f};
    for (int c = 0; c < NCHUNK; ++c, ++t) {
      if (w == 0) {
        if (t + 1 < TOTAL_CHUNKS) {
          // next chunk: row kt2*16, col chunk c2*CHUNK_M of padded W
          const int t1 = t + 1;
          const int kt2 = t1 / NCHUNK;
          const int c2 = t1 - kt2 * NCHUNK;
          tdm_issue_w((unsigned)(WOFF + (t1 & 1) * WBUF_BYTES),
                      wbf + (size_t)kt2 * 16 * M_PROTO + c2 * CHUNK_M);
          __builtin_amdgcn_s_wait_tensorcnt(1);  // in-order: chunk t landed
        } else {
          __builtin_amdgcn_s_wait_tensorcnt(0);
        }
      }
      __syncthreads();  // W chunk t visible to all waves
      const char* wb = wstage + (t & 1) * WBUF_BYTES;
#pragma unroll 4
      for (int mm = 0; mm < CHUNK_M; mm += 32) {
        const int m0 = c * CHUNK_M + mm;
        v16bf a = load_a_frag(arow + m0 + half * 8);  // D from LDS
        v16bf b = load_a_frag((const __bf16*)(wb + l16 * WROW_BYTES) + mm + half * 8);
        acc = wmma_bf16(a, b, acc);
      }
      __syncthreads();  // protect buffer (t&1) before it is re-filled at t+2
    }
#pragma unroll
    for (int i = 0; i < 8; ++i) {
      float cc = acc[i];
      float cm = valid ? cc : -3.0e38f;
      float nm = fmaxf(mrun[i], cm);
      srun[i] = srun[i] * __expf(mrun[i] - nm) + (valid ? __expf(cc - nm) : 0.0f);
      mrun[i] = nm;
      if (valid && kcol == yv[i]) ly[i] += cc;  // gather logit at y[row]
    }
  }
  // combine per-lane logsumexp states across the 16 column lanes of each half
#pragma unroll
  for (int i = 0; i < 8; ++i) {
#pragma unroll
    for (int off = 1; off < 16; off <<= 1) {
      float om = __shfl_xor(mrun[i], off, 16);
      float os = __shfl_xor(srun[i], off, 16);
      float nm = fmaxf(mrun[i], om);
      srun[i] = srun[i] * __expf(mrun[i] - nm) + os * __expf(om - nm);
      mrun[i] = nm;
      ly[i] += __shfl_xor(ly[i], off, 16);
    }
  }
  if (l16 == 0) {
    float s = 0.f;
#pragma unroll
    for (int i = 0; i < 8; ++i) s += mrun[i] + __logf(srun[i]) - ly[i];
    atomicAdd(&accum[0], s);  // sum over rows of (logsumexp - logit_y)
  }
}

// ---------------- final reduction ----------------
__global__ __launch_bounds__(256) void k_final(const float* __restrict__ r1min,
                                               const float* __restrict__ accum,
                                               const float* __restrict__ l1p,
                                               const float* __restrict__ l2p,
                                               float* __restrict__ out) {
  __shared__ float red[256];
  float s = 0.f;
  for (int c = threadIdx.x; c < M_PROTO; c += 256) s += r1min[c];
  red[threadIdx.x] = s;
  __syncthreads();
  for (int st = 128; st > 0; st >>= 1) {
    if (threadIdx.x < st) red[threadIdx.x] += red[threadIdx.x + st];
    __syncthreads();
  }
  if (threadIdx.x == 0) {
    const float inv_n = 1.0f / (float)N_SAMP;
    out[0] = accum[0] * inv_n + l1p[0] * (red[0] / (float)M_PROTO) +
             l2p[0] * accum[1] * inv_n;
  }
}

// ---------------- launcher ----------------
extern "C" void kernel_launch(void* const* d_in, const int* in_sizes, int n_in,
                              void* d_out, int out_size, void* d_ws, size_t ws_size,
                              hipStream_t stream) {
  const float* x = (const float*)d_in[0];      // [N,P]
  const float* proto = (const float*)d_in[1];  // [M,P]
  const float* W = (const float*)d_in[2];      // [K,M]
  const int* y = (const int*)d_in[3];          // [N]
  const float* l1 = (const float*)d_in[4];
  const float* l2 = (const float*)d_in[5];
  float* out = (float*)d_out;

  // workspace carve-up (~40 MB)
  char* ws = (char*)d_ws;
  __bf16* xbf = (__bf16*)ws;  ws += (size_t)N_SAMP * P_DIM * sizeof(__bf16);
  __bf16* pbf = (__bf16*)ws;  ws += (size_t)M_PROTO * P_DIM * sizeof(__bf16);
  __bf16* wbf = (__bf16*)ws;  ws += (size_t)K_PAD * M_PROTO * sizeof(__bf16);
  float* x2 = (float*)ws;     ws += (size_t)N_SAMP * sizeof(float);
  float* p2 = (float*)ws;     ws += (size_t)M_PROTO * sizeof(float);
  float* r1min = (float*)ws;  ws += (size_t)M_PROTO * sizeof(float);
  float* accum = (float*)ws;  ws += 256;

  k_init<<<8, 256, 0, stream>>>(r1min, accum);
  k_row_prep<<<N_SAMP / 8, 256, 0, stream>>>(x, xbf, x2, N_SAMP, P_DIM);
  k_row_prep<<<M_PROTO / 8, 256, 0, stream>>>(proto, pbf, p2, M_PROTO, P_DIM);
  k_w_prep<<<(K_PAD * M_PROTO + 255) / 256, 256, 0, stream>>>(W, wbf);

  k_main<<<N_SAMP / ROWS_PER_BLK, 128, SMEM_TOTAL, stream>>>(xbf, pbf, wbf, x2,
                                                             p2, y, r1min, accum);
  k_final<<<1, 256, 0, stream>>>(r1min, accum, l1, l2, out);
}
